// NonLocalBlock2D_68917045232007
// MI455X (gfx1250) — matrix operand order repaired and checked by reading, not compiled, not run on hardware
//
#include <hip/hip_runtime.h>

typedef __bf16 bf16;
typedef __attribute__((ext_vector_type(16))) __bf16 v16bf;
typedef __attribute__((ext_vector_type(8)))  float  v8f;
typedef __attribute__((ext_vector_type(4)))  unsigned int u32x4;
typedef __attribute__((ext_vector_type(8)))  int i32x8;
typedef __attribute__((ext_vector_type(4)))  int i32x4;

#define C_IN  256
#define C_INT 128
#define BATCH 8
#define NPOS  4096          // 64*64
#define NT    (NPOS / 16)   // 256 position tiles
#define WPB   8             // waves per attention block
#define NITER (NPOS / 32)   // 128 column steps

#if defined(__gfx1250__) && __has_builtin(__builtin_amdgcn_tensor_load_to_lds)
#define USE_TDM 1
#else
#define USE_TDM 0
#endif

// ---------- WMMA fragment helpers (contiguous k = 16*half + s order, wave32) ----------

// 16-bit A-matrix 16x32: lanes 0-15 row M=lane hold K=0..15,
// lanes 16-31 row M=lane-16 hold K=16..31 (matches B). 2 elems per VGPR.
__device__ __forceinline__ int a_koff(int j, int half) {
  return half * 16 + j * 2;
}
// 16-bit B-matrix 32x16: lane = column; lanes 0-15 hold K=0..15, lanes 16-31 K=16..31.
__device__ __forceinline__ int b_koff(int j, int half) {
  return half * 16 + j * 2;
}

__device__ __forceinline__ v16bf loadA_f32(const float* __restrict__ row, int half) {
  v16bf f;
#pragma unroll
  for (int j = 0; j < 8; ++j) {
    int k = a_koff(j, half);
    f[2 * j]     = (bf16)row[k];
    f[2 * j + 1] = (bf16)row[k + 1];
  }
  return f;
}
__device__ __forceinline__ v16bf loadA_bf16(const bf16* row, int half) {
  v16bf f;
#pragma unroll
  for (int j = 0; j < 8; ++j) {
    int k = a_koff(j, half);
    f[2 * j]     = row[k];
    f[2 * j + 1] = row[k + 1];
  }
  return f;
}
__device__ __forceinline__ v16bf loadB_f32_strided(const float* __restrict__ col,
                                                   long kstride, int half) {
  v16bf f;
#pragma unroll
  for (int j = 0; j < 8; ++j) {
    int k = b_koff(j, half);
    f[2 * j]     = (bf16)col[(long)k * kstride];
    f[2 * j + 1] = (bf16)col[(long)(k + 1) * kstride];
  }
  return f;
}
// K-pairs contiguous within the half's 16-element run -> wide loads
__device__ __forceinline__ v16bf loadB_bf16(const bf16* col, int half) {
  v16bf f;
#pragma unroll
  for (int j = 0; j < 8; ++j) {
    int k = b_koff(j, half);
    f[2 * j]     = col[k];
    f[2 * j + 1] = col[k + 1];
  }
  return f;
}

__device__ __forceinline__ v8f wmma_bf16(v16bf a, v16bf b, v8f c) {
  return __builtin_amdgcn_wmma_f32_16x16x32_bf16(false, a, false, b, (short)0, c,
                                                 false, false);
}

__device__ __forceinline__ float red_max16(float v) {
#pragma unroll
  for (int m = 1; m < 16; m <<= 1) v = fmaxf(v, __shfl_xor(v, m));
  return v;
}
__device__ __forceinline__ float red_sum16(float v) {
#pragma unroll
  for (int m = 1; m < 16; m <<= 1) v += __shfl_xor(v, m);
  return v;
}

#if USE_TDM
// Generic LDS pointer: aperture lives in addr[63:32], low 32 bits = LDS byte addr
__device__ __forceinline__ unsigned lds_off(const void* p) {
  return (unsigned)(unsigned long long)p;
}
// Issue a 2D TDM tile load (bf16 elements) per CDNA5 ISA D# layout (ch.8).
// Groups 2/3 zero: <=2D tensor. Tracked by TENSORcnt.
// This toolchain exposes the 6-arg builtin: (g0, g1, g2, g3, g_extra, cpol).
__device__ __forceinline__ void tdm_load_2d(unsigned lds_addr,
                                            unsigned long long gaddr,
                                            unsigned tensor_d0, unsigned tensor_d1,
                                            unsigned tile_d0, unsigned tile_d1,
                                            unsigned long long d0_stride) {
  u32x4 g0;
  g0[0] = 1u;                                        // count=1 (valid), user mode
  g0[1] = lds_addr;                                  // lds_addr[31:0]
  g0[2] = (unsigned)(gaddr & 0xffffffffull);         // global_addr[31:0]
  g0[3] = (unsigned)((gaddr >> 32) & 0x1ffffffull)   // global_addr[56:32]
          | (2u << 30);                              // type = 2 ("image")
  i32x8 g1;
  g1[0] = (int)(1u << 16);                           // wg_mask=0, data_size=1 (2B)
  g1[1] = (int)((tensor_d0 & 0xffffu) << 16);        // [63:48] tensor_dim0 lo16
  g1[2] = (int)(((tensor_d0 >> 16) & 0xffffu) |      // [79:64] tensor_dim0 hi16
                ((tensor_d1 & 0xffffu) << 16));      // [95:80] tensor_dim1 lo16
  g1[3] = (int)(((tensor_d1 >> 16) & 0xffffu) |      // [111:96] tensor_dim1 hi16
                ((tile_d0 & 0xffffu) << 16));        // [127:112] tile_dim0
  g1[4] = (int)(tile_d1 & 0xffffu);                  // tile_dim1 (tile_dim2 = 0)
  g1[5] = (int)(d0_stride & 0xffffffffull);          // dim0_stride[31:0]
  g1[6] = (int)((d0_stride >> 32) & 0xffffull);      // dim0_stride[47:32]
  g1[7] = 0;                                         // dim1_stride (unused, 2D)
  i32x4 z4 = {0, 0, 0, 0};
  i32x8 z8 = {0, 0, 0, 0, 0, 0, 0, 0};
  __builtin_amdgcn_tensor_load_to_lds(g0, g1, z4, z4, z8, 0);
}
#endif

// ---------------- Kernel 1: three 1x1-conv projections -----------------
// theta_t, phi_t: [B, N, 128] bf16 (position-major), g_cm: [B, 128, N] bf16
__global__ void nl_proj_kernel(const float* __restrict__ x,
                               const float* __restrict__ g_w, const float* __restrict__ g_b,
                               const float* __restrict__ th_w, const float* __restrict__ th_b,
                               const float* __restrict__ ph_w, const float* __restrict__ ph_b,
                               bf16* __restrict__ theta_t, bf16* __restrict__ phi_t,
                               bf16* __restrict__ g_cm) {
  const int lane = threadIdx.x, half = lane >> 4, l16 = lane & 15;
  const int b = blockIdx.x / NT, nt = blockIdx.x % NT;
  const int n0 = nt * 16, wid = blockIdx.y;  // 0=theta 1=phi 2=g
  const float* W  = (wid == 0) ? th_w : (wid == 1) ? ph_w : g_w;
  const float* Bv = (wid == 0) ? th_b : (wid == 1) ? ph_b : g_b;

  v8f acc[8];
#pragma unroll
  for (int ct = 0; ct < 8; ++ct) acc[ct] = (v8f){0, 0, 0, 0, 0, 0, 0, 0};

  const int n = n0 + l16;
#pragma unroll
  for (int ks = 0; ks < C_IN / 32; ++ks) {
    v16bf bt = loadB_f32_strided(x + ((long)b * C_IN + ks * 32) * NPOS + n, NPOS, half);
#pragma unroll
    for (int ct = 0; ct < 8; ++ct) {
      v16bf at = loadA_f32(W + (long)(ct * 16 + l16) * C_IN + ks * 32, half);
      acc[ct] = wmma_bf16(at, bt, acc[ct]);
    }
  }
#pragma unroll
  for (int ct = 0; ct < 8; ++ct) {
#pragma unroll
    for (int r = 0; r < 8; ++r) {
      const int c = ct * 16 + r + 8 * half;
      const float v = acc[ct][r] + Bv[c];
      if (wid == 2)
        g_cm[((long)b * C_INT + c) * NPOS + n] = (bf16)v;
      else if (wid == 0)
        theta_t[((long)b * NPOS + n) * C_INT + c] = (bf16)v;
      else
        phi_t[((long)b * NPOS + n) * C_INT + c] = (bf16)v;
    }
  }
}

// -------- Kernel 2: fused flash attention (scores -> softmax -> PV) -----
// 8 waves per block share double-buffered phi/g tiles in LDS. On TDM
// toolchains, wave 0 issues tensor_load_to_lds descriptors and the DMA
// engine streams global->LDS concurrently with the WMMAs (TENSORcnt);
// otherwise a register-staged cooperative copy is used.
__global__ __launch_bounds__(32 * WPB)
void nl_attn_kernel(const bf16* __restrict__ theta_t,
                    const bf16* __restrict__ phi_t,
                    const bf16* __restrict__ g_cm,
                    bf16* __restrict__ y_t) {
  __shared__ __align__(16) bf16 lphi[2][32 * C_INT];  // [pos_local][ch]  16 KB
  __shared__ __align__(16) bf16 lg[2][C_INT * 32];    // [ch][pos_local]  16 KB
  __shared__ __align__(16) bf16 lp[WPB][16 * 32];     // per-wave P tiles  8 KB

  const int tid  = threadIdx.x;
  const int w    = tid >> 5, lane = tid & 31;
  const int half = (lane >> 4) & 1, l16 = lane & 15;
  const int b    = blockIdx.x >> 5;        // 32 tile-groups per batch
  const int grp  = blockIdx.x & 31;
  const int i0   = (grp * WPB + w) * 16;   // this wave's query rows

  // Query A-fragments for this wave's 16 rows, K = 128 channels (4 x 32)
  v16bf qa[4];
  {
    const bf16* row = theta_t + ((long)b * NPOS + i0 + l16) * C_INT;
#pragma unroll
    for (int kk = 0; kk < 4; ++kk) qa[kk] = loadA_bf16(row + kk * 32, half);
  }

  v8f o[8];
#pragma unroll
  for (int ct = 0; ct < 8; ++ct) o[ct] = (v8f){0, 0, 0, 0, 0, 0, 0, 0};
  float mrow[8], lrow[8];
#pragma unroll
  for (int r = 0; r < 8; ++r) { mrow[r] = -__builtin_inff(); lrow[r] = 0.f; }

  const bf16* phib = phi_t + (long)b * NPOS * C_INT;
  const bf16* gb_  = g_cm + (long)b * C_INT * NPOS;

#if USE_TDM
  // phi tile: 32 rows x 128 bf16, row stride 128 (contiguous block)
  // g tile:   128 rows x 32 bf16, row stride 4096
  if (w == 0) {
    tdm_load_2d(lds_off(lphi[0]), (unsigned long long)(phib),
                C_INT, 32, C_INT, 32, C_INT);
    tdm_load_2d(lds_off(lg[0]), (unsigned long long)(gb_),
                32, C_INT, 32, C_INT, NPOS);
    __builtin_amdgcn_s_wait_tensorcnt(0);
  }
  __syncthreads();

  for (int jt = 0; jt < NITER; ++jt) {
    const int cur = jt & 1;
    if (w == 0 && jt + 1 < NITER) {  // async DMA of next tile, overlaps WMMAs
      const int jp = (jt + 1) * 32;
      tdm_load_2d(lds_off(lphi[cur ^ 1]),
                  (unsigned long long)(phib + (long)jp * C_INT),
                  C_INT, 32, C_INT, 32, C_INT);
      tdm_load_2d(lds_off(lg[cur ^ 1]),
                  (unsigned long long)(gb_ + jp),
                  32, C_INT, 32, C_INT, NPOS);
    }
#else
  const u32x4* phiv = (const u32x4*)phib;
  const u32x4* gv   = (const u32x4*)gb_;
  const int q1 = tid + 256;
  u32x4 rp0, rp1, rg0, rg1;
  // stage tile 0
  rp0 = phiv[((tid >> 4)) * 16 + (tid & 15)];
  rp1 = phiv[((q1 >> 4)) * 16 + (q1 & 15)];
  rg0 = gv[(tid >> 2) * 512 + (tid & 3)];
  rg1 = gv[(q1 >> 2) * 512 + (q1 & 3)];
  ((u32x4*)lphi[0])[tid] = rp0; ((u32x4*)lphi[0])[q1] = rp1;
  ((u32x4*)lg[0])[tid]   = rg0; ((u32x4*)lg[0])[q1]   = rg1;
  __syncthreads();

  for (int jt = 0; jt < NITER; ++jt) {
    const int cur = jt & 1;
    if (jt + 1 < NITER) {  // next tile's global loads overlap compute
      const int jp = (jt + 1) * 32, j4 = (jt + 1) * 4;
      rp0 = phiv[(jp + (tid >> 4)) * 16 + (tid & 15)];
      rp1 = phiv[(jp + (q1 >> 4)) * 16 + (q1 & 15)];
      rg0 = gv[(tid >> 2) * 512 + j4 + (tid & 3)];
      rg1 = gv[(q1 >> 2) * 512 + j4 + (q1 & 3)];
    }
#endif
    if (jt + 2 < NITER) {  // global_prefetch_b8 two steps ahead
      const int jp2 = (jt + 2) * 32;
      __builtin_prefetch(phib + ((long)jp2 + (tid >> 3)) * C_INT, 0, 0);
      __builtin_prefetch(gb_ + (long)(tid >> 1) * NPOS + jp2, 0, 0);
    }

    const bf16* lph = lphi[cur];
    const bf16* lgc = lg[cur];

    // ---- scores S = theta x phi^T for 16x32 tile, fed from LDS ----
    v8f s0 = (v8f){0, 0, 0, 0, 0, 0, 0, 0};
    v8f s1 = (v8f){0, 0, 0, 0, 0, 0, 0, 0};
#pragma unroll
    for (int kk = 0; kk < 4; ++kk) {
      v16bf b0 = loadB_bf16(lph + l16 * C_INT + kk * 32, half);
      s0 = wmma_bf16(qa[kk], b0, s0);
      v16bf b1 = loadB_bf16(lph + (16 + l16) * C_INT + kk * 32, half);
      s1 = wmma_bf16(qa[kk], b1, s1);
    }

    // ---- online softmax: lane group owns rows r + 8*half (C layout) ----
    float alpha[8];
#pragma unroll
    for (int r = 0; r < 8; ++r) {
      float mx   = red_max16(fmaxf(s0[r], s1[r]));
      float mnew = fmaxf(mrow[r], mx);
      float p0 = __expf(s0[r] - mnew);
      float p1 = __expf(s1[r] - mnew);
      float rs = red_sum16(p0 + p1);
      alpha[r] = __expf(mrow[r] - mnew);
      lrow[r]  = lrow[r] * alpha[r] + rs;
      mrow[r]  = mnew;
      s0[r] = p0; s1[r] = p1;
    }
#pragma unroll
    for (int ct = 0; ct < 8; ++ct)
#pragma unroll
      for (int r = 0; r < 8; ++r) o[ct][r] *= alpha[r];

    // ---- P tile C-layout -> A-layout via per-wave LDS slice ----
    // (DS ops of one wave execute in order: store->load RAW is safe)
    bf16* lpw = lp[w];
#pragma unroll
    for (int r = 0; r < 8; ++r) {
      const int row = r + 8 * half;
      lpw[row * 32 + l16]      = (bf16)s0[r];
      lpw[row * 32 + 16 + l16] = (bf16)s1[r];
    }
    v16bf pa = loadA_bf16(lpw + l16 * 32, half);

    // ---- O[16x128] += P[16x32] x G[32x128], G fed from LDS ----
#pragma unroll
    for (int ct = 0; ct < 8; ++ct) {
      v16bf gfrag = loadB_bf16(lgc + (ct * 16 + l16) * 32, half);
      o[ct] = wmma_bf16(pa, gfrag, o[ct]);
    }

#if USE_TDM
    if (w == 0 && jt + 1 < NITER) __builtin_amdgcn_s_wait_tensorcnt(0);
    __syncthreads();  // next buffer complete + everyone done with cur
  }
#else
    if (jt + 1 < NITER) {  // publish next buffer (nobody reads it during jt)
      ((u32x4*)lphi[cur ^ 1])[tid] = rp0; ((u32x4*)lphi[cur ^ 1])[q1] = rp1;
      ((u32x4*)lg[cur ^ 1])[tid]   = rg0; ((u32x4*)lg[cur ^ 1])[q1]   = rg1;
    }
    __syncthreads();
  }
#endif

  // Normalize and store y tile position-major for the mask GEMM
#pragma unroll
  for (int r = 0; r < 8; ++r) {
    const float inv = 1.0f / lrow[r];
    const int pos = i0 + r + 8 * half;
#pragma unroll
    for (int ct = 0; ct < 8; ++ct)
      y_t[((long)b * NPOS + pos) * C_INT + ct * 16 + l16] = (bf16)(o[ct][r] * inv);
  }
}

// ---------- Kernel 3: mask 1x1 conv (256x128 GEMM) + residual ----------
__global__ void nl_mask_kernel(const float* __restrict__ x,
                               const float* __restrict__ mask_w,
                               const float* __restrict__ mask_b,
                               const bf16* __restrict__ y_t,
                               float* __restrict__ out) {
  const int lane = threadIdx.x, half = lane >> 4, l16 = lane & 15;
  const int b = blockIdx.x / NT, nt = blockIdx.x % NT;
  const int n0 = nt * 16, c0 = blockIdx.y * 16;
  const int n = n0 + l16;

  v8f acc = (v8f){0, 0, 0, 0, 0, 0, 0, 0};
#pragma unroll
  for (int kk = 0; kk < C_INT / 32; ++kk) {
    v16bf at = loadA_f32(mask_w + (long)(c0 + l16) * C_INT + kk * 32, half);
    v16bf bt = loadB_bf16(y_t + ((long)b * NPOS + n) * C_INT + kk * 32, half);
    acc = wmma_bf16(at, bt, acc);
  }
#pragma unroll
  for (int r = 0; r < 8; ++r) {
    const int c = c0 + r + 8 * half;
    const long idx = ((long)b * C_IN + c) * NPOS + n;
    out[idx] = acc[r] + mask_b[c] + x[idx];
  }
}

extern "C" void kernel_launch(void* const* d_in, const int* in_sizes, int n_in,
                              void* d_out, int out_size, void* d_ws, size_t ws_size,
                              hipStream_t stream) {
  (void)in_sizes; (void)n_in; (void)out_size; (void)ws_size;
  const float* x    = (const float*)d_in[0];
  const float* g_w  = (const float*)d_in[1];
  const float* g_b  = (const float*)d_in[2];
  const float* th_w = (const float*)d_in[3];
  const float* th_b = (const float*)d_in[4];
  const float* ph_w = (const float*)d_in[5];
  const float* ph_b = (const float*)d_in[6];
  const float* mk_w = (const float*)d_in[7];
  const float* mk_b = (const float*)d_in[8];
  float* out = (float*)d_out;

  const size_t TSZ = (size_t)BATCH * NPOS * C_INT;  // 4 Mi elems (8 MB bf16)
  bf16* theta_t = (bf16*)d_ws;
  bf16* phi_t   = theta_t + TSZ;
  bf16* g_cm    = phi_t + TSZ;
  bf16* y_t     = g_cm + TSZ;      // total 32 MB of workspace

  dim3 wave(32, 1, 1);
  nl_proj_kernel<<<dim3(BATCH * NT, 3), wave, 0, stream>>>(
      x, g_w, g_b, th_w, th_b, ph_w, ph_b, theta_t, phi_t, g_cm);
  nl_attn_kernel<<<dim3(BATCH * 32), dim3(32 * WPB), 0, stream>>>(
      theta_t, phi_t, g_cm, y_t);
  nl_mask_kernel<<<dim3(BATCH * NT, C_IN / 16), wave, 0, stream>>>(
      x, mk_w, mk_b, y_t, out);
}
